// TiledFeedForward_6691559047432
// MI455X (gfx1250) — matrix operand order repaired and checked by reading, not compiled, not run on hardware
//
#include <hip/hip_runtime.h>

// ---------------------------------------------------------------------------
// CDNA5 (gfx1250) fused 2-layer MLP:  out = relu(X*W1 + b1)*W2 + b2
// bf16 WMMA (v_wmma_f32_16x16x32_bf16, fp32 accum) with double-buffered LDS
// tiles filled by GLOBAL_LOAD_ASYNC_TO_LDS_B128 (ASYNCcnt-tracked async copy).
// ---------------------------------------------------------------------------

typedef __attribute__((ext_vector_type(16))) __bf16 v16bf;
typedef __attribute__((ext_vector_type(8)))  __bf16 v8bf;
typedef __attribute__((ext_vector_type(4)))  __bf16 v4bf;
typedef __attribute__((ext_vector_type(8)))  float  v8f;

#define HIDDEN 1024
#define INTER  4096
#define MTOT   (4 * 2048)   // 8192 rows

#define BM 128
#define BN 128
#define BK 32
#define PAD 8                 // bf16 elements of row padding (16 bytes)
#define LROW (BK + PAD)       // LDS row stride in elements (80 bytes)

union Frag16 { v16bf v; v8bf h[2]; };

// Async global->LDS 16-byte copy. Per ISA 10.2, the low 32 bits of a flat
// LDS address are the LDS byte offset, which is exactly the VDST operand of
// global_load_async_to_lds_b128. Tracked by ASYNCcnt.
__device__ __forceinline__ void async_copy_b128(void* lds_dst, const void* gsrc) {
  unsigned lofs = (unsigned)(unsigned long long)lds_dst;       // LDS byte offset
  unsigned long long ga = (unsigned long long)gsrc;            // 64-bit global addr
  asm volatile("global_load_async_to_lds_b128 %0, %1, off"
               :: "v"(lofs), "v"(ga)
               : "memory");
}

__device__ __forceinline__ void wait_async_all() {
  asm volatile("s_wait_asynccnt 0x0" ::: "memory");
}

// --------------------------- fp32 -> bf16 (flat) ---------------------------
__global__ __launch_bounds__(256) void convert_f32_bf16(
    const float* __restrict__ in, __bf16* __restrict__ out, int n4) {
  int i = blockIdx.x * 256 + threadIdx.x;
  if (i < n4) {
    float4 f = ((const float4*)in)[i];
    v4bf o;
    o[0] = (__bf16)f.x; o[1] = (__bf16)f.y; o[2] = (__bf16)f.z; o[3] = (__bf16)f.w;
    ((v4bf*)out)[i] = o;
  }
}

// ------------------- fp32 [R,C] -> bf16 transposed [C,R] -------------------
__global__ __launch_bounds__(256) void transpose_convert(
    const float* __restrict__ in, __bf16* __restrict__ out, int R, int C) {
  __shared__ __bf16 t[32][33];
  const int tx = threadIdx.x;          // 0..31
  const int ty = threadIdx.y;          // 0..7
  const int r0 = blockIdx.y * 32;
  const int c0 = blockIdx.x * 32;
#pragma unroll
  for (int j = 0; j < 4; ++j)
    t[ty + j * 8][tx] = (__bf16)in[(long)(r0 + ty + j * 8) * C + c0 + tx];
  __syncthreads();
#pragma unroll
  for (int j = 0; j < 4; ++j)
    out[(long)(c0 + ty + j * 8) * R + r0 + tx] = t[tx][ty + j * 8];
}

// ------------------------------ tiled GEMM ---------------------------------
// A:  [M,K] bf16 row-major (K contiguous)
// BT: [N,K] bf16 row-major (K contiguous) == B transposed
// RELU_BF16: epilogue = relu(acc+bias) stored bf16, else acc+bias stored f32.
template <bool RELU_BF16>
__global__ __launch_bounds__(256) void mlp_gemm(
    const __bf16* __restrict__ A, const __bf16* __restrict__ BT,
    const float* __restrict__ bias, void* __restrict__ outp,
    int M, int N, int K) {
  __shared__ __attribute__((aligned(16))) __bf16 As[2][BM][LROW];
  __shared__ __attribute__((aligned(16))) __bf16 Bs[2][BN][LROW];

  const int tid  = threadIdx.x;
  const int lane = tid & 31;
  const int wid  = tid >> 5;      // 8 wave32s
  const int wm   = wid & 3;       // 4 waves along M -> 32 rows each
  const int wn   = wid >> 2;      // 2 waves along N -> 64 cols each
  const int m0   = blockIdx.y * BM;
  const int n0   = blockIdx.x * BN;

  const int lrow = lane & 15;     // row (A) / col (B) within 16-tile
  const int kg   = lane >> 4;     // K half-group select

  // Per-thread async-copy assignment: 4 x B128 per tile (2 A rows + 2 B rows),
  // branch-free. row = tid>>2 (0..63) and +64; vec = (tid&3)*8 elements.
  const int crow4 = tid >> 2;
  const int cvec  = (tid & 3) * 8;
  const __bf16* gA0 = A  + (long)(m0 + crow4)      * K + cvec;
  const __bf16* gA1 = A  + (long)(m0 + crow4 + 64) * K + cvec;
  const __bf16* gB0 = BT + (long)(n0 + crow4)      * K + cvec;
  const __bf16* gB1 = BT + (long)(n0 + crow4 + 64) * K + cvec;

  v8f acc[2][4];
#pragma unroll
  for (int im = 0; im < 2; ++im)
#pragma unroll
    for (int jn = 0; jn < 4; ++jn)
      acc[im][jn] = (v8f){0.f,0.f,0.f,0.f,0.f,0.f,0.f,0.f};

  // issue async loads for a K-slab into LDS buffer `buf`
  auto load_tile = [&](int buf, int kb) {
    async_copy_b128(&As[buf][crow4][cvec],      gA0 + kb);
    async_copy_b128(&As[buf][crow4 + 64][cvec], gA1 + kb);
    async_copy_b128(&Bs[buf][crow4][cvec],      gB0 + kb);
    async_copy_b128(&Bs[buf][crow4 + 64][cvec], gB1 + kb);
  };

  load_tile(0, 0);  // prologue: tile 0 in flight

  for (int kb = 0, t = 0; kb < K; kb += BK, ++t) {
    const int buf = t & 1;

    // tile t resident in LDS + all waves finished reading buf^1
    wait_async_all();
    __syncthreads();

    if (kb + BK < K) load_tile(buf ^ 1, kb + BK);  // overlap with WMMA below

    // ---- fragment loads per ISA 16-bit A(16x32) / B(32x16) layouts ----
    Frag16 a[2], b[4];
#pragma unroll
    for (int im = 0; im < 2; ++im) {
      const __bf16* p = &As[buf][wm * 32 + im * 16 + lrow][0];
      a[im].h[0] = *(const v8bf*)(p + kg * 8);        // K = kg*8 .. +7
      a[im].h[1] = *(const v8bf*)(p + 16 + kg * 8);   // K = 16+kg*8 .. +7
    }
#pragma unroll
    for (int jn = 0; jn < 4; ++jn) {
      const __bf16* p = &Bs[buf][wn * 64 + jn * 16 + lrow][0];
      b[jn].h[0] = *(const v8bf*)(p + kg * 8);        // K = kg*8 .. +7
      b[jn].h[1] = *(const v8bf*)(p + 16 + kg * 8);   // K = 16+kg*8 .. +7
    }

#pragma unroll
    for (int im = 0; im < 2; ++im)
#pragma unroll
      for (int jn = 0; jn < 4; ++jn)
        acc[im][jn] = __builtin_amdgcn_wmma_f32_16x16x32_bf16(
            false, a[im].v, false, b[jn].v, (short)0, acc[im][jn], false, false);
  }

  // ---- epilogue: C/D layout -> VGPR r holds M=(lane>=16?8:0)+r, N=lane&15 --
  const int erow = (lane >> 4) * 8;
  const int ecol = lane & 15;
#pragma unroll
  for (int im = 0; im < 2; ++im) {
#pragma unroll
    for (int jn = 0; jn < 4; ++jn) {
      const int n  = n0 + wn * 64 + jn * 16 + ecol;
      const float bv = bias[n];
#pragma unroll
      for (int r = 0; r < 8; ++r) {
        const int m = m0 + wm * 32 + im * 16 + erow + r;
        float v = acc[im][jn][r] + bv;
        if (RELU_BF16) {
          v = v > 0.f ? v : 0.f;
          ((__bf16*)outp)[(long)m * N + n] = (__bf16)v;
        } else {
          ((float*)outp)[(long)m * N + n] = v;
        }
      }
    }
  }
}

// ------------------------------- launcher ----------------------------------
extern "C" void kernel_launch(void* const* d_in, const int* in_sizes, int n_in,
                              void* d_out, int out_size, void* d_ws, size_t ws_size,
                              hipStream_t stream) {
  const float* X  = (const float*)d_in[0];  // [4,2048,1024]
  const float* W1 = (const float*)d_in[1];  // [1024,4096]  (in,out)
  const float* b1 = (const float*)d_in[2];  // [4096]
  const float* W2 = (const float*)d_in[3];  // [4096,1024]
  const float* b2 = (const float*)d_in[4];  // [1024]
  float* out = (float*)d_out;               // [4,2048,1024] f32

  char* ws = (char*)d_ws;
  __bf16* Xbf = (__bf16*)ws;  ws += (size_t)MTOT  * HIDDEN * 2;  //  16 MB
  __bf16* W1T = (__bf16*)ws;  ws += (size_t)HIDDEN * INTER * 2;  // 8.4 MB [INTER,HIDDEN]
  __bf16* W2T = (__bf16*)ws;  ws += (size_t)HIDDEN * INTER * 2;  // 8.4 MB [HIDDEN,INTER]
  __bf16* Hbf = (__bf16*)ws;                                     //  67 MB [MTOT,INTER]

  // conversion prepasses
  {
    int n4 = MTOT * HIDDEN / 4;
    convert_f32_bf16<<<(n4 + 255) / 256, 256, 0, stream>>>(X, Xbf, n4);
  }
  transpose_convert<<<dim3(INTER / 32, HIDDEN / 32), dim3(32, 8), 0, stream>>>(
      W1, W1T, HIDDEN, INTER);
  transpose_convert<<<dim3(HIDDEN / 32, INTER / 32), dim3(32, 8), 0, stream>>>(
      W2, W2T, INTER, HIDDEN);

  // GEMM1: H = relu(X*W1 + b1)   -> bf16  [8192, 4096]
  mlp_gemm<true><<<dim3(INTER / BN, MTOT / BM), 256, 0, stream>>>(
      Xbf, W1T, b1, (void*)Hbf, MTOT, INTER, HIDDEN);

  // GEMM2: out = H*W2 + b2       -> f32   [8192, 1024]
  mlp_gemm<false><<<dim3(HIDDEN / BN, MTOT / BM), 256, 0, stream>>>(
      Hbf, W2T, b2, (void*)out, MTOT, HIDDEN, INTER);
}
